// TransformerProjector_63556926046955
// MI455X (gfx1250) — compile-verified, matrix operand order checked by reading, not run on hardware
//
#include <hip/hip_runtime.h>
#include <hip/hip_bf16.h>

// ---------------------------------------------------------------------------
// CDNA5 (gfx1250) bf16 WMMA transformer projector.
// wave32; v_wmma_f32_16x16x32_bf16 for all GEMM / attention contractions.
// ---------------------------------------------------------------------------

typedef float  v8f   __attribute__((ext_vector_type(8)));
typedef __bf16 v16bf __attribute__((ext_vector_type(16)));
typedef __bf16 v4bf  __attribute__((ext_vector_type(4)));

static __device__ __forceinline__ v8f wmma_bf16(v16bf a, v16bf b, v8f c) {
  // D = A(16x32) x B(32x16) + C(16x16), f32 accumulate
  return __builtin_amdgcn_wmma_f32_16x16x32_bf16(
      /*neg_a=*/false, a, /*neg_b=*/false, b,
      /*c_mod=*/(short)0, c, /*reuse_a=*/false, /*reuse_b=*/false);
}

// ---------------------------------------------------------------------------
// fp32 -> bf16 conversion (weights, once per launch)
// ---------------------------------------------------------------------------
__global__ void cvt_bf16_kernel(const float* __restrict__ src,
                                __bf16* __restrict__ dst, int n) {
  int i = blockIdx.x * 256 + threadIdx.x;
  if (i < n) dst[i] = (__bf16)src[i];
}

// ---------------------------------------------------------------------------
// plane [512][4096] -> enc_in [4096][512]
// ---------------------------------------------------------------------------
__global__ void transpose_cs_kernel(const float* __restrict__ src,
                                    float* __restrict__ dst) {
  int i = blockIdx.x * 256 + threadIdx.x;      // over 4096*512
  int s = i >> 9, c = i & 511;
  dst[(size_t)s * 512 + c] = src[(size_t)c * 4096 + s];
}

// ---------------------------------------------------------------------------
// trilinear grid-sample of pos_enc[512][32][32][32] at query points
// ---------------------------------------------------------------------------
__global__ __launch_bounds__(256) void grid_sample_kernel(
    const float* __restrict__ vol, const float* __restrict__ pts,
    float* __restrict__ out) {
  int p = blockIdx.x;
  float gx = pts[p * 3 + 0], gy = pts[p * 3 + 1], gz = pts[p * 3 + 2];
  float ix = ((gx + 1.f) * 32.f - 1.f) * 0.5f;
  float iy = ((gy + 1.f) * 32.f - 1.f) * 0.5f;
  float iz = ((gz + 1.f) * 32.f - 1.f) * 0.5f;
  float x0 = floorf(ix), y0 = floorf(iy), z0 = floorf(iz);
  float fx = ix - x0, fy = iy - y0, fz = iz - z0;
  int tid = threadIdx.x;
  for (int cc = 0; cc < 2; ++cc) {
    int c = tid + cc * 256;
    float accv = 0.f;
#pragma unroll
    for (int dz = 0; dz < 2; ++dz)
#pragma unroll
      for (int dy = 0; dy < 2; ++dy)
#pragma unroll
        for (int dx = 0; dx < 2; ++dx) {
          float xi = x0 + dx, yi = y0 + dy, zi = z0 + dz;
          float wgt = (dx ? fx : 1.f - fx) * (dy ? fy : 1.f - fy) *
                      (dz ? fz : 1.f - fz);
          bool ok = (xi >= 0.f) && (xi < 32.f) && (yi >= 0.f) && (yi < 32.f) &&
                    (zi >= 0.f) && (zi < 32.f);
          int xc = (int)fminf(fmaxf(xi, 0.f), 31.f);
          int yc = (int)fminf(fmaxf(yi, 0.f), 31.f);
          int zc = (int)fminf(fmaxf(zi, 0.f), 31.f);
          accv += vol[((size_t)c * 32 + zc) * 1024 + yc * 32 + xc] *
                  (ok ? wgt : 0.f);
        }
    out[(size_t)p * 512 + c] = accv;
  }
}

// ---------------------------------------------------------------------------
// residual add + LayerNorm over D=512, one row per 256-thread block
// ---------------------------------------------------------------------------
__global__ __launch_bounds__(256) void ln512_kernel(
    const float* __restrict__ x, const float* __restrict__ res,
    const float* __restrict__ g, const float* __restrict__ b,
    float* __restrict__ out) {
  const int row = blockIdx.x;
  const int tid = threadIdx.x;
  __shared__ float red[256];
  const float* xr = x + (size_t)row * 512;
  const float* rr = res + (size_t)row * 512;
  float v0 = xr[tid] + rr[tid];
  float v1 = xr[tid + 256] + rr[tid + 256];
  red[tid] = v0 + v1;
  __syncthreads();
  for (int o = 128; o > 0; o >>= 1) {
    if (tid < o) red[tid] += red[tid + o];
    __syncthreads();
  }
  float mean = red[0] * (1.0f / 512.0f);
  __syncthreads();
  float d0 = v0 - mean, d1 = v1 - mean;
  red[tid] = d0 * d0 + d1 * d1;
  __syncthreads();
  for (int o = 128; o > 0; o >>= 1) {
    if (tid < o) red[tid] += red[tid + o];
    __syncthreads();
  }
  float rstd = rsqrtf(red[0] * (1.0f / 512.0f) + 1e-5f);
  out[(size_t)row * 512 + tid] = d0 * rstd * g[tid] + b[tid];
  out[(size_t)row * 512 + tid + 256] = d1 * rstd * g[tid + 256] + b[tid + 256];
}

// ---------------------------------------------------------------------------
// pack [S][512] f32 -> [8][S][64] bf16 (optionally scaled, for Q/K)
// ---------------------------------------------------------------------------
__global__ void pack_heads_kernel(const float* __restrict__ src,
                                  __bf16* __restrict__ dst, int S, float scale) {
  int i = blockIdx.x * 256 + threadIdx.x;
  if (i >= S * 512) return;
  int s = i >> 9, c = i & 511;
  int hh = c >> 6, d = c & 63;
  dst[((size_t)hh * S + s) * 64 + d] = (__bf16)(src[i] * scale);
}

// pack [S][512] f32 -> Vt [8][64][S] bf16 (per-head transpose)
__global__ void pack_vT_kernel(const float* __restrict__ src,
                               __bf16* __restrict__ dst, int S) {
  int i = blockIdx.x * 256 + threadIdx.x;
  if (i >= S * 512) return;
  int s = i >> 9, c = i & 511;
  int hh = c >> 6, d = c & 63;
  dst[((size_t)hh * 64 + d) * S + s] = (__bf16)src[i];
}

// ---------------------------------------------------------------------------
// WMMA GEMM:  C[M][N] = act( A[M][K](f32) * W[N][K](bf16)^T + bias[N] )
// Block tile 128x64, BK=64, 8 waves of 32x32 (2x2 WMMA tiles, 8 WMMA/stage).
// Software-pipelined: next k-stage's global loads issued right after the
// barrier so they overlap the current stage's WMMA block.
// act: 0 = none, 1 = exact-erf GELU
// ---------------------------------------------------------------------------
__global__ __launch_bounds__(256) void gemm_xwt_kernel(
    const float* __restrict__ A, const __bf16* __restrict__ W,
    const float* __restrict__ bias, float* __restrict__ C,
    int M, int N, int K, int act) {
  __shared__ __align__(32) __bf16 As[128 * 64];   // 16 KB
  __shared__ __align__(32) __bf16 Ws[64 * 64];    //  8 KB
  const int tid = threadIdx.x;
  const int m0 = blockIdx.x * 128;
  const int n0 = blockIdx.y * 64;
  const int lane = tid & 31;
  const int wv = tid >> 5;
  const int wm = (wv & 3) * 32;
  const int wn = (wv >> 2) * 32;
  const int col = lane & 15;
  const int hf = lane >> 4;

  float4 ra[8];                              // in-flight A tile (128x64 f32)
  uint2  rw[4];                              // in-flight W tile (64x64 bf16)
  auto load_stage = [&](int k0) {
#pragma unroll
    for (int i = 0; i < 8; ++i) {
      int base = (i * 256 + tid) * 4;
      ra[i] = *(const float4*)&A[(size_t)(m0 + (base >> 6)) * K + k0 + (base & 63)];
    }
#pragma unroll
    for (int i = 0; i < 4; ++i) {
      int base = (i * 256 + tid) * 4;
      rw[i] = *(const uint2*)&W[(size_t)(n0 + (base >> 6)) * K + k0 + (base & 63)];
    }
  };

  v8f a00 = {}, a01 = {}, a10 = {}, a11 = {};
  load_stage(0);

  for (int k0 = 0; k0 < K; k0 += 64) {
#pragma unroll
    for (int i = 0; i < 8; ++i) {            // pack f32x4 -> bf16x4, b64 store
      int base = (i * 256 + tid) * 4;
      v4bf pk = { (__bf16)ra[i].x, (__bf16)ra[i].y,
                  (__bf16)ra[i].z, (__bf16)ra[i].w };
      *(v4bf*)&As[base] = pk;
    }
#pragma unroll
    for (int i = 0; i < 4; ++i) {
      int base = (i * 256 + tid) * 4;
      *(uint2*)&Ws[base] = rw[i];
    }
    __syncthreads();
    if (k0 + 64 < K) load_stage(k0 + 64);    // overlap next loads with WMMA
#pragma unroll
    for (int kk = 0; kk < 2; ++kk) {
      v16bf af0 = *(const v16bf*)&As[(wm + col) * 64 + kk * 32 + hf * 16];
      v16bf af1 = *(const v16bf*)&As[(wm + 16 + col) * 64 + kk * 32 + hf * 16];
      v16bf wf0 = *(const v16bf*)&Ws[(wn + col) * 64 + kk * 32 + hf * 16];
      v16bf wf1 = *(const v16bf*)&Ws[(wn + 16 + col) * 64 + kk * 32 + hf * 16];
      a00 = wmma_bf16(af0, wf0, a00);
      a01 = wmma_bf16(af0, wf1, a01);
      a10 = wmma_bf16(af1, wf0, a10);
      a11 = wmma_bf16(af1, wf1, a11);
    }
    __syncthreads();
  }

  const float* bptr = bias + n0;
  const float bias0 = bptr[wn + col];        // sn = 0 column
  const float bias1 = bptr[wn + 16 + col];   // sn = 1 column
#pragma unroll
  for (int sm = 0; sm < 2; ++sm) {
#pragma unroll
    for (int sn = 0; sn < 2; ++sn) {
      v8f a = (sm == 0) ? (sn == 0 ? a00 : a01) : (sn == 0 ? a10 : a11);
      float bb = sn == 0 ? bias0 : bias1;
#pragma unroll
      for (int r = 0; r < 8; ++r) {
        int mm = wm + sm * 16 + r + hf * 8;
        int nn = wn + sn * 16 + col;
        float v = a[r] + bb;
        if (act) v = 0.5f * v * (1.0f + erff(v * 0.70710678118654752f));
        C[(size_t)(m0 + mm) * N + n0 + nn] = v;
      }
    }
  }
}

// ---------------------------------------------------------------------------
// Flash attention, one wave per (head, 16-query tile).
//  qh [8][Sq][64] bf16 (pre-scaled by Dh^-0.5), kh [8][Sk][64] bf16,
//  vT [8][64][Sk] bf16.  out [Sq][512] f32 (heads merged).
// Transposed-score trick: sT = K*Q^T so softmax reductions are in-lane
// plus one shfl_xor(16); then O^T += V^T * P.
// mode: 0 = dense, 1 = causal (k<=q), 2 = banded |k-q|<=band
// ---------------------------------------------------------------------------
__global__ __launch_bounds__(32) void flash_attn_kernel(
    const __bf16* __restrict__ qh, const __bf16* __restrict__ kh,
    const __bf16* __restrict__ vT, float* __restrict__ out,
    int Sq, int Sk, int mode, int band) {
  const int qtiles = Sq >> 4;
  const int head = blockIdx.x / qtiles;
  const int q0 = (blockIdx.x % qtiles) << 4;
  const int lane = threadIdx.x & 31;
  const int col = lane & 15;   // query column / matrix N index
  const int hf = lane >> 4;    // half-wave

  // Q as B-fragment: lane holds column q=q0+col, d-range hf*16 (+32)
  const __bf16* qbase = qh + ((size_t)head * Sq + q0 + col) * 64 + hf * 16;
  v16bf qb0 = *(const v16bf*)(qbase);
  v16bf qb1 = *(const v16bf*)(qbase + 32);

  v8f acc[4] = {};                 // O^T, 4 blocks of 16 d-rows x 16 q-cols
  float mrun = -3.0e38f, lrun = 0.f;

  int klo = 0, khi = Sk;
  if (mode == 1) khi = q0 + 16;
  if (mode == 2) {
    klo = q0 - band; if (klo < 0) klo = 0;
    khi = q0 + 16 + band; if (khi > Sk) khi = Sk;
  }
  klo &= ~31;

  const __bf16* vbase = vT + (size_t)head * 64 * Sk;

  for (int kt = klo; kt < khi; kt += 32) {
    const __bf16* kbase = kh + ((size_t)head * Sk + kt) * 64;
    v8f s0 = {}, s1 = {};          // sT for key sub-tiles [kt..+15], [+16..+31]
    {
      v16bf ka;
      ka = *(const v16bf*)(kbase + (size_t)col * 64 + hf * 16);
      s0 = wmma_bf16(ka, qb0, s0);
      ka = *(const v16bf*)(kbase + (size_t)col * 64 + 32 + hf * 16);
      s0 = wmma_bf16(ka, qb1, s0);
      ka = *(const v16bf*)(kbase + (size_t)(col + 16) * 64 + hf * 16);
      s1 = wmma_bf16(ka, qb0, s1);
      ka = *(const v16bf*)(kbase + (size_t)(col + 16) * 64 + 32 + hf * 16);
      s1 = wmma_bf16(ka, qb1, s1);
    }
    const int qabs = q0 + col;
    float e0[8], e1[8];
    float tmax = -3.0e38f;
#pragma unroll
    for (int r = 0; r < 8; ++r) {
      int k0a = kt + r + hf * 8;       // C layout: row m = r (+8 upper half)
      int k1a = k0a + 16;
      float v0s = s0[r], v1s = s1[r];
      bool ok0 = true, ok1 = true;
      if (mode == 1) { ok0 = (k0a <= qabs); ok1 = (k1a <= qabs); }
      if (mode == 2) {
        int dd0 = k0a - qabs; ok0 = (dd0 <= band) && (-dd0 <= band);
        int dd1 = k1a - qabs; ok1 = (dd1 <= band) && (-dd1 <= band);
      }
      v0s = ok0 ? v0s : -3.0e38f;
      v1s = ok1 ? v1s : -3.0e38f;
      e0[r] = v0s; e1[r] = v1s;
      tmax = fmaxf(tmax, fmaxf(v0s, v1s));
    }
    tmax = fmaxf(tmax, __shfl_xor(tmax, 16, 32));
    float mnew = fmaxf(mrun, tmax);
    float alpha = __expf(mrun - mnew);
    float tsum = 0.f;
#pragma unroll
    for (int r = 0; r < 8; ++r) {
      e0[r] = __expf(e0[r] - mnew);
      e1[r] = __expf(e1[r] - mnew);
      tsum += e0[r] + e1[r];
    }
    tsum += __shfl_xor(tsum, 16, 32);
    lrun = lrun * alpha + tsum;
    mrun = mnew;
#pragma unroll
    for (int i = 0; i < 4; ++i)
#pragma unroll
      for (int r = 0; r < 8; ++r) acc[i][r] *= alpha;

    // build P as B-fragment: lane (col,hf) needs K = hf*16 .. hf*16+15;
    // the complementary 8 k-rows live in the partner lane (xor 16).
    float f0[8], f1[8];
#pragma unroll
    for (int r = 0; r < 8; ++r) {
      f0[r] = __shfl_xor(e0[r], 16, 32);
      f1[r] = __shfl_xor(e1[r], 16, 32);
    }
    v16bf pb;
#pragma unroll
    for (int r = 0; r < 8; ++r) {
      float lo = hf ? f1[r] : e0[r];
      float hi = hf ? e1[r] : f0[r];
      pb[r] = (__bf16)lo;
      pb[r + 8] = (__bf16)hi;
    }
    // O^T += V^T * P   (A = V^T tile, contiguous loads thanks to vT layout)
#pragma unroll
    for (int i = 0; i < 4; ++i) {
      v16bf va = *(const v16bf*)(vbase + (size_t)(i * 16 + col) * Sk + kt + hf * 16);
      acc[i] = wmma_bf16(va, pb, acc[i]);
    }
  }

  float inv = 1.0f / lrun;
#pragma unroll
  for (int i = 0; i < 4; ++i)
#pragma unroll
    for (int r = 0; r < 8; ++r) {
      int d = i * 16 + r + hf * 8;
      out[(size_t)(q0 + col) * 512 + head * 64 + d] = acc[i][r] * inv;
    }
}

// ---------------------------------------------------------------------------
// host orchestration
// ---------------------------------------------------------------------------
extern "C" void kernel_launch(void* const* d_in, const int* in_sizes, int n_in,
                              void* d_out, int out_size, void* d_ws,
                              size_t ws_size, hipStream_t stream) {
  (void)in_sizes; (void)n_in; (void)out_size; (void)ws_size;
  const float* plane     = (const float*)d_in[0];
  const float* query     = (const float*)d_in[1];
  const float* posenc    = (const float*)d_in[2];
  const float* enc_pos   = (const float*)d_in[3];
  const float* dec_pos   = (const float*)d_in[4];
  const float* enc_eln_w = (const float*)d_in[5];
  const float* enc_eln_b = (const float*)d_in[6];
  const float* dec_eln_w = (const float*)d_in[7];
  const float* dec_eln_b = (const float*)d_in[8];
  const float* enc_attn_w = (const float*)d_in[9];
  const float* enc_attn_b = (const float*)d_in[10];
  const float* enc_fc1_w  = (const float*)d_in[11];
  const float* enc_fc1_b  = (const float*)d_in[12];
  const float* enc_fc2_w  = (const float*)d_in[13];
  const float* enc_fc2_b  = (const float*)d_in[14];
  const float* enc_ln_w   = (const float*)d_in[15];
  const float* enc_ln_b   = (const float*)d_in[16];
  const float* dec_attn_w = (const float*)d_in[17];
  const float* dec_attn_b = (const float*)d_in[18];
  const float* dec_fc1_w  = (const float*)d_in[19];
  const float* dec_fc1_b  = (const float*)d_in[20];
  const float* dec_fc2_w  = (const float*)d_in[21];
  const float* dec_fc2_b  = (const float*)d_in[22];
  const float* dec_ln_w   = (const float*)d_in[23];
  const float* dec_ln_b   = (const float*)d_in[24];

  const int S = 4096, Nq = 1024, D = 512, F = 1024, H = 8;

  char* wp = (char*)d_ws;
  auto carve = [&](size_t bytes) -> void* {
    void* r = (void*)wp;
    wp += (bytes + 255) & ~(size_t)255;
    return r;
  };
  __bf16* wEncAttn = (__bf16*)carve((size_t)6 * 4 * D * D * 2);
  __bf16* wEncF1   = (__bf16*)carve((size_t)6 * F * D * 2);
  __bf16* wEncF2   = (__bf16*)carve((size_t)6 * D * F * 2);
  __bf16* wDecAttn = (__bf16*)carve((size_t)6 * 8 * D * D * 2);
  __bf16* wDecF1   = (__bf16*)carve((size_t)6 * F * D * 2);
  __bf16* wDecF2   = (__bf16*)carve((size_t)6 * D * F * 2);
  float* hEnc = (float*)carve((size_t)S * D * 4);
  float* hDec = (float*)carve((size_t)Nq * D * 4);
  float* bQ   = (float*)carve((size_t)S * D * 4);
  float* bK   = (float*)carve((size_t)S * D * 4);
  float* bV   = (float*)carve((size_t)S * D * 4);
  float* bA   = (float*)carve((size_t)S * D * 4);
  float* bF   = (float*)carve((size_t)S * F * 4);
  __bf16* qh  = (__bf16*)carve((size_t)H * S * 64 * 2);
  __bf16* kh  = (__bf16*)carve((size_t)H * S * 64 * 2);
  __bf16* vTh = (__bf16*)carve((size_t)H * 64 * S * 2);

  auto cvt = [&](const float* src, __bf16* dst, long long n) {
    cvt_bf16_kernel<<<(unsigned)((n + 255) / 256), 256, 0, stream>>>(src, dst, (int)n);
  };
  cvt(enc_attn_w, wEncAttn, 6LL * 4 * D * D);
  cvt(enc_fc1_w,  wEncF1,   6LL * F * D);
  cvt(enc_fc2_w,  wEncF2,   6LL * D * F);
  cvt(dec_attn_w, wDecAttn, 6LL * 8 * D * D);
  cvt(dec_fc1_w,  wDecF1,   6LL * F * D);
  cvt(dec_fc2_w,  wDecF2,   6LL * D * F);

  auto gemm = [&](const float* A, const __bf16* W, const float* bias, float* C,
                  int M, int N, int K, int act) {
    dim3 grid(M / 128, N / 64);
    gemm_xwt_kernel<<<grid, 256, 0, stream>>>(A, W, bias, C, M, N, K, act);
  };
  auto ln = [&](const float* x, const float* res, const float* g,
                const float* b, float* out, int rows) {
    ln512_kernel<<<rows, 256, 0, stream>>>(x, res, g, b, out);
  };
  auto packqk = [&](const float* s, __bf16* d2, int SS, float sc) {
    pack_heads_kernel<<<SS * 2, 256, 0, stream>>>(s, d2, SS, sc);
  };
  auto packv = [&](const float* s, __bf16* d2, int SS) {
    pack_vT_kernel<<<SS * 2, 256, 0, stream>>>(s, d2, SS);
  };
  auto attn = [&](float* O, int Sq, int Sk, int mode) {
    flash_attn_kernel<<<H * (Sq / 16), 32, 0, stream>>>(qh, kh, vTh, O, Sq, Sk,
                                                        mode, 512);
  };

  // ---------------- encoder ----------------
  transpose_cs_kernel<<<(S * D) / 256, 256, 0, stream>>>(plane, bA);
  ln(bA, enc_pos, enc_eln_w, enc_eln_b, hEnc, S);
  for (int i = 0; i < 6; ++i) {
    const __bf16* Wq = wEncAttn + (size_t)(i * 4 + 0) * D * D;
    const __bf16* Wk = wEncAttn + (size_t)(i * 4 + 1) * D * D;
    const __bf16* Wv = wEncAttn + (size_t)(i * 4 + 2) * D * D;
    const __bf16* Wo = wEncAttn + (size_t)(i * 4 + 3) * D * D;
    const float* bq = enc_attn_b + (size_t)(i * 4 + 0) * D;
    const float* bk = enc_attn_b + (size_t)(i * 4 + 1) * D;
    const float* bv = enc_attn_b + (size_t)(i * 4 + 2) * D;
    const float* bo = enc_attn_b + (size_t)(i * 4 + 3) * D;
    gemm(hEnc, Wq, bq, bQ, S, D, D, 0);
    gemm(hEnc, Wk, bk, bK, S, D, D, 0);
    gemm(hEnc, Wv, bv, bV, S, D, D, 0);
    packqk(bQ, qh, S, 0.125f);
    packqk(bK, kh, S, 1.0f);
    packv(bV, vTh, S);
    attn(bQ, S, S, /*banded*/2);
    gemm(bQ, Wo, bo, bA, S, D, D, 0);
    ln(hEnc, bA, enc_ln_w + (size_t)(i * 2 + 0) * D,
       enc_ln_b + (size_t)(i * 2 + 0) * D, hEnc, S);
    gemm(hEnc, wEncF1 + (size_t)i * F * D, enc_fc1_b + (size_t)i * F, bF,
         S, F, D, /*gelu*/1);
    gemm(bF, wEncF2 + (size_t)i * D * F, enc_fc2_b + (size_t)i * D, bA,
         S, D, F, 0);
    ln(hEnc, bA, enc_ln_w + (size_t)(i * 2 + 1) * D,
       enc_ln_b + (size_t)(i * 2 + 1) * D, hEnc, S);
  }

  // ---------------- decoder ----------------
  grid_sample_kernel<<<Nq, 256, 0, stream>>>(posenc, query, bA);
  ln(bA, dec_pos, dec_eln_w, dec_eln_b, hDec, Nq);
  for (int i = 0; i < 6; ++i) {
    const __bf16* Wb = wDecAttn + (size_t)i * 8 * D * D;
    const float* Bb = dec_attn_b + (size_t)i * 8 * D;
    // causal self-attention
    gemm(hDec, Wb + 0 * D * D, Bb + 0 * D, bQ, Nq, D, D, 0);
    gemm(hDec, Wb + 1 * D * D, Bb + 1 * D, bK, Nq, D, D, 0);
    gemm(hDec, Wb + 2 * D * D, Bb + 2 * D, bV, Nq, D, D, 0);
    packqk(bQ, qh, Nq, 0.125f);
    packqk(bK, kh, Nq, 1.0f);
    packv(bV, vTh, Nq);
    attn(bQ, Nq, Nq, /*causal*/1);
    gemm(bQ, Wb + 3 * D * D, Bb + 3 * D, bA, Nq, D, D, 0);
    ln(hDec, bA, dec_ln_w + (size_t)(i * 3 + 0) * D,
       dec_ln_b + (size_t)(i * 3 + 0) * D, hDec, Nq);
    // cross-attention to encoder output
    gemm(hDec, Wb + 4 * D * D, Bb + 4 * D, bQ, Nq, D, D, 0);
    gemm(hEnc, Wb + 5 * D * D, Bb + 5 * D, bK, S, D, D, 0);
    gemm(hEnc, Wb + 6 * D * D, Bb + 6 * D, bV, S, D, D, 0);
    packqk(bQ, qh, Nq, 0.125f);
    packqk(bK, kh, S, 1.0f);
    packv(bV, vTh, S);
    attn(bQ, Nq, S, /*dense*/0);
    gemm(bQ, Wb + 7 * D * D, Bb + 7 * D, bA, Nq, D, D, 0);
    ln(hDec, bA, dec_ln_w + (size_t)(i * 3 + 1) * D,
       dec_ln_b + (size_t)(i * 3 + 1) * D, hDec, Nq);
    // FFN
    gemm(hDec, wDecF1 + (size_t)i * F * D, dec_fc1_b + (size_t)i * F, bF,
         Nq, F, D, /*gelu*/1);
    gemm(bF, wDecF2 + (size_t)i * D * F, dec_fc2_b + (size_t)i * D, bA,
         Nq, D, F, 0);
    ln(hDec, bA, dec_ln_w + (size_t)(i * 3 + 2) * D,
       dec_ln_b + (size_t)(i * 3 + 2) * D, hDec, Nq);
  }

  hipMemcpyAsync(d_out, hDec, (size_t)Nq * D * sizeof(float),
                 hipMemcpyDeviceToDevice, stream);
}
